// JacobianLayer_88630945120273
// MI455X (gfx1250) — compile-verified, matrix operand order checked by reading, not run on hardware
//
#include <hip/hip_runtime.h>

// Problem sizes (fixed by the reference).
#define BATCH 8192
#define DD 64     // state dim (Jacobian is DD x DD per batch element)
#define HH 256    // hidden dim

typedef __attribute__((ext_vector_type(16))) _Float16 v16h;
typedef __attribute__((ext_vector_type(8)))  float    v8f;

// ---------------------------------------------------------------------------
// Prep: f32 weights -> f16, with W2/W3 transposed so WMMA B-fragments
// (lane = output column N, vector elements = K) read contiguous memory.
//   w1h[i*256+m] = W1[i,m]            (64x256, row-major, A-operand of GEMM1)
//   w2t[j*256+m] = W2[m,j]            (256x256, B-operand of GEMM1)
//   w3t[k*256+j] = W3[j,k]            (64x256,  B-operand of GEMM2)
// ---------------------------------------------------------------------------
__global__ __launch_bounds__(256) void jac_prep_kernel(
    const float* __restrict__ W1, const float* __restrict__ W2,
    const float* __restrict__ W3,
    _Float16* __restrict__ w1h, _Float16* __restrict__ w2t,
    _Float16* __restrict__ w3t)
{
    int idx = blockIdx.x * 256 + threadIdx.x;
    if (idx < HH * HH) {                       // W2 transpose
        int j = idx >> 8, m = idx & 255;
        w2t[idx] = (_Float16)W2[m * HH + j];
    } else if (idx < HH * HH + DD * HH) {      // W3 transpose
        int t = idx - HH * HH;
        int k = t >> 8, j = t & 255;
        w3t[t] = (_Float16)W3[j * DD + k];
    } else if (idx < HH * HH + 2 * DD * HH) {  // W1 convert
        int t = idx - HH * HH - DD * HH;
        w1h[t] = (_Float16)W1[t];
    }
}

// ---------------------------------------------------------------------------
// Main kernel: one workgroup (8 wave32) per batch element.
//   Forward pass (scalar, per hidden unit)  ->  g1, g2 in LDS
//   GEMM1 (WMMA f16): U = W1 @ (diag(g1) W2), V = U ∘ g2  -> LDS (f16)
//   GEMM2 (WMMA f16): Jt = V @ W3, write jac[b,k,i] = Jt[i,k]
// Fragment layout per CDNA5 ISA (16-bit A 16x32): lane L holds row M=L%16;
// element e maps to K = (e<8 ? e : e+8) + (L>=16 ? 8 : 0) within a 32-chunk.
// B mirrors this with lane = column N. Each wave owns two adjacent N-tiles
// and shares one A-fragment across both (2 WMMAs per A load).
// ---------------------------------------------------------------------------
__global__ __launch_bounds__(256) void jac_main_kernel(
    const float* __restrict__ x,  const float* __restrict__ W1,
    const float* __restrict__ b1, const float* __restrict__ W2,
    const float* __restrict__ b2,
    const _Float16* __restrict__ w1h, const _Float16* __restrict__ w2t,
    const _Float16* __restrict__ w3t,
    float* __restrict__ out)
{
    __shared__ float    xs[DD];
    __shared__ float    h1s[HH];
    __shared__ _Float16 g1h[HH];
    __shared__ float    g2s[HH];
    __shared__ _Float16 V[DD * HH];   // 32 KB: g2-scaled intermediate, f16

    const int tid = threadIdx.x;
    const int b   = blockIdx.x;

    // ---- forward pass: one hidden unit per thread ----
    if (tid < DD) xs[tid] = x[b * DD + tid];
    __syncthreads();

    float a1 = b1[tid];
    #pragma unroll 8
    for (int d = 0; d < DD; ++d) a1 += xs[d] * W1[d * HH + tid];
    float h1 = tanhf(a1);
    h1s[tid] = h1;
    g1h[tid] = (_Float16)(1.0f - h1 * h1);
    __syncthreads();

    float a2 = b2[tid];
    #pragma unroll 8
    for (int m = 0; m < HH; ++m) a2 += h1s[m] * W2[m * HH + tid];
    float h2 = tanhf(a2);
    g2s[tid] = 1.0f - h2 * h2;
    __syncthreads();

    const int wave  = tid >> 5;
    const int lane  = tid & 31;
    const int row16 = lane & 15;   // M (A) or N (B, D) index within tile
    const int hi    = lane >> 4;   // selects K sub-block / M+8 rows of D

    // ---- GEMM1: U = W1f16 @ (diag(g1) W2); 64x256 output, 16x16 tiles ----
    // Wave w owns N-tiles {2w, 2w+1}; one A-fragment feeds both (2 WMMAs).
    {
        const int j0 = (wave * 2) * 16 + row16;     // lane column, N-tile 2w
        const int j1 = j0 + 16;                     // lane column, N-tile 2w+1
        v8f acc0[4] = {};
        v8f acc1[4] = {};
        #pragma unroll 2
        for (int tk = 0; tk < 8; ++tk) {            // K = 256 in chunks of 32
            // B fragments: columns j0/j1 of W2 (pre-transposed), scaled by g1[K]
            v16h bf0, bf1;
            const _Float16* bp0 = w2t + j0 * HH + tk * 32 + hi * 8;
            const _Float16* bp1 = w2t + j1 * HH + tk * 32 + hi * 8;
            const _Float16* gp  = g1h + tk * 32 + hi * 8;
            #pragma unroll
            for (int e = 0; e < 8; ++e) {
                _Float16 g = gp[e];
                bf0[e] = bp0[e] * g;
                bf1[e] = bp1[e] * g;
            }
            #pragma unroll
            for (int e = 0; e < 8; ++e) {
                _Float16 g = gp[16 + e];
                bf0[8 + e] = bp0[16 + e] * g;
                bf1[8 + e] = bp1[16 + e] * g;
            }
            #pragma unroll
            for (int ti = 0; ti < 4; ++ti) {
                v16h af;
                const _Float16* ap = w1h + (ti * 16 + row16) * HH + tk * 32 + hi * 8;
                #pragma unroll
                for (int e = 0; e < 8; ++e) af[e]     = ap[e];
                #pragma unroll
                for (int e = 0; e < 8; ++e) af[8 + e] = ap[16 + e];
                acc0[ti] = __builtin_amdgcn_wmma_f32_16x16x32_f16(
                    false, af, false, bf0, (short)0, acc0[ti], false, false);
                acc1[ti] = __builtin_amdgcn_wmma_f32_16x16x32_f16(
                    false, af, false, bf1, (short)0, acc1[ti], false, false);
            }
        }
        // scale by g2[j], store f16 into LDS V (D layout: row = ti*16+hi*8+r)
        const float g2v0 = g2s[j0];
        const float g2v1 = g2s[j1];
        #pragma unroll
        for (int ti = 0; ti < 4; ++ti) {
            #pragma unroll
            for (int r = 0; r < 8; ++r) {
                const int row = ti * 16 + hi * 8 + r;
                V[row * HH + j0] = (_Float16)(acc0[ti][r] * g2v0);
                V[row * HH + j1] = (_Float16)(acc1[ti][r] * g2v1);
            }
        }
    }
    __syncthreads();

    // ---- GEMM2: Jt = V @ W3 (64x64); 16 tiles, 2 per wave (same M tile) ----
    {
        const int t0  = wave * 2;
        const int ti  = t0 >> 2;                 // M tile 0..3 (input dim i)
        const int k0  = (t0 & 3) * 16 + row16;   // output column, tile t0
        const int k1  = k0 + 16;                 // output column, tile t0+1
        v8f acc0 = {};
        v8f acc1 = {};
        #pragma unroll
        for (int tk = 0; tk < 8; ++tk) {
            v16h af, bf0, bf1;
            const _Float16* ap  = V   + (ti * 16 + row16) * HH + tk * 32 + hi * 8;
            const _Float16* bp0 = w3t + k0 * HH + tk * 32 + hi * 8;
            const _Float16* bp1 = w3t + k1 * HH + tk * 32 + hi * 8;
            #pragma unroll
            for (int e = 0; e < 8; ++e) {
                af[e] = ap[e]; bf0[e] = bp0[e]; bf1[e] = bp1[e];
            }
            #pragma unroll
            for (int e = 0; e < 8; ++e) {
                af[8 + e] = ap[16 + e]; bf0[8 + e] = bp0[16 + e]; bf1[8 + e] = bp1[16 + e];
            }
            acc0 = __builtin_amdgcn_wmma_f32_16x16x32_f16(
                false, af, false, bf0, (short)0, acc0, false, false);
            acc1 = __builtin_amdgcn_wmma_f32_16x16x32_f16(
                false, af, false, bf1, (short)0, acc1, false, false);
        }
        // jac[b, k, i] = Jt[i, k]; rows i = ti*16 + hi*8 + r are contiguous
        float* o0 = out + (size_t)b * DD * DD + (size_t)k0 * DD + ti * 16 + hi * 8;
        float* o1 = out + (size_t)b * DD * DD + (size_t)k1 * DD + ti * 16 + hi * 8;
        #pragma unroll
        for (int r = 0; r < 8; ++r) o0[r] = acc0[r];
        #pragma unroll
        for (int r = 0; r < 8; ++r) o1[r] = acc1[r];
    }
}

extern "C" void kernel_launch(void* const* d_in, const int* in_sizes, int n_in,
                              void* d_out, int out_size, void* d_ws, size_t ws_size,
                              hipStream_t stream) {
    (void)in_sizes; (void)n_in; (void)out_size; (void)ws_size;
    const float* x  = (const float*)d_in[0];
    const float* W1 = (const float*)d_in[1];
    const float* b1 = (const float*)d_in[2];
    const float* W2 = (const float*)d_in[3];
    const float* b2 = (const float*)d_in[4];
    const float* W3 = (const float*)d_in[5];
    // d_in[6] = b3: constant offset, vanishes in the Jacobian.

    _Float16* w2t = (_Float16*)d_ws;         // 65536 halves (128 KB)
    _Float16* w3t = w2t + HH * HH;           // 16384 halves (32 KB)
    _Float16* w1h = w3t + DD * HH;           // 16384 halves (32 KB)

    const int prep_elems  = HH * HH + 2 * DD * HH;   // 98304
    jac_prep_kernel<<<(prep_elems + 255) / 256, 256, 0, stream>>>(
        W1, W2, W3, w1h, w2t, w3t);
    jac_main_kernel<<<BATCH, 256, 0, stream>>>(
        x, W1, b1, W2, b2, w1h, w2t, w3t, (float*)d_out);
}
